// DeepTopK_44684839747849
// MI455X (gfx1250) — matrix operand order repaired and checked by reading, not compiled, not run on hardware
//
#include <hip/hip_runtime.h>
#include <hip/hip_bf16.h>

typedef __attribute__((ext_vector_type(16))) __bf16 v16bf;
typedef __attribute__((ext_vector_type(8)))  float  v8f;
typedef __attribute__((ext_vector_type(4)))  float  f32x4;
typedef __attribute__((ext_vector_type(4)))  unsigned int u32x4;
typedef __attribute__((ext_vector_type(2)))  unsigned int u32x2;

#define BM 64
#define BN 128
#define BK 64
#define NTHREADS 256
#define LDK 72          // BK + 8 pad (ushorts): 144B row stride -> conflict-free ds_load_b128
#define NBINS 2048

__device__ __forceinline__ unsigned short f32_to_bf16_rne(float f) {
  unsigned u = __float_as_uint(f);
  u += 0x7FFFu + ((u >> 16) & 1u);
  return (unsigned short)(u >> 16);
}
__device__ __forceinline__ float bf16_hi_f32(unsigned short h) {
  return __uint_as_float(((unsigned)h) << 16);
}

union FragBF16 { u32x4 q[2]; v16bf v; };

// C = act(A @ B + bias); A-side threshold mask fused on load; value histogram fused in epilogue.
__global__ __launch_bounds__(NTHREADS)
void gemm_bias_act_hist_kernel(const float* __restrict__ A,
                               const float* __restrict__ B,
                               const float* __restrict__ bias,
                               float* __restrict__ C,
                               int M, int N, int K,
                               const float* __restrict__ thresh,  // nullable
                               unsigned* __restrict__ hist,       // nullable
                               int relu)
{
  __shared__ unsigned short AsHi[BM * LDK];
  __shared__ unsigned short AsLo[BM * LDK];
  __shared__ unsigned short BsHi[BN * LDK];
  __shared__ unsigned short BsLo[BN * LDK];
  __shared__ unsigned shHist[NBINS];

  const int tid   = threadIdx.x;
  const int lane  = tid & 31;
  const int wid   = tid >> 5;
  const int waveM = wid & 1;     // 2 waves along M
  const int waveN = wid >> 1;    // 4 waves along N
  const int bm = blockIdx.y * BM;
  const int bn = blockIdx.x * BN;

  if (hist) {
    for (int i = tid; i < NBINS; i += NTHREADS) shHist[i] = 0;
  }

  const float th = thresh ? *thresh : -3.0e38f;

  f32x4 regA[4];   // 64x64/4 f4 / 256 thr = 4
  f32x4 regB[8];   // 64x128/4 f4 / 256 thr = 8
  const int kTiles = K / BK;

  auto loadA = [&](int kt) {
    const float* Ab = A + (size_t)bm * K + (size_t)kt * BK;
#pragma unroll
    for (int j = 0; j < 4; ++j) {
      int i = tid + NTHREADS * j;
      int r = i >> 4, c4 = i & 15;              // 16 float4 per 64-wide row
      f32x4 v = *(const f32x4*)(Ab + (size_t)r * K + c4 * 4);
#pragma unroll
      for (int e = 0; e < 4; ++e) v[e] = (v[e] >= th) ? v[e] : 0.0f;
      regA[j] = v;
    }
  };
  auto loadB = [&](int kt) {
    const float* Bb = B + (size_t)(kt * BK) * N + bn;
#pragma unroll
    for (int j = 0; j < 8; ++j) {
      int i = tid + NTHREADS * j;
      int r = i >> 5, c4 = i & 31;              // 32 float4 per 128-wide row
      regB[j] = *(const f32x4*)(Bb + (size_t)r * N + c4 * 4);
    }
  };
  auto storeA = [&]() {
#pragma unroll
    for (int j = 0; j < 4; ++j) {
      int i = tid + NTHREADS * j;
      int r = i >> 4, c4 = i & 15;
      f32x4 v = regA[j];
      unsigned short h[4], l[4];
#pragma unroll
      for (int e = 0; e < 4; ++e) {
        h[e] = f32_to_bf16_rne(v[e]);
        l[e] = f32_to_bf16_rne(v[e] - bf16_hi_f32(h[e]));
      }
      u32x2 ph, pl;
      ph[0] = (unsigned)h[0] | ((unsigned)h[1] << 16);
      ph[1] = (unsigned)h[2] | ((unsigned)h[3] << 16);
      pl[0] = (unsigned)l[0] | ((unsigned)l[1] << 16);
      pl[1] = (unsigned)l[2] | ((unsigned)l[3] << 16);
      *(u32x2*)&AsHi[r * LDK + c4 * 4] = ph;
      *(u32x2*)&AsLo[r * LDK + c4 * 4] = pl;
    }
  };
  auto storeB = [&]() {   // transpose [k][n] -> Bs[n][k] while splitting hi/lo
#pragma unroll
    for (int j = 0; j < 8; ++j) {
      int i = tid + NTHREADS * j;
      int r = i >> 5, c4 = i & 31;
      f32x4 v = regB[j];
#pragma unroll
      for (int e = 0; e < 4; ++e) {
        unsigned short h = f32_to_bf16_rne(v[e]);
        unsigned short l = f32_to_bf16_rne(v[e] - bf16_hi_f32(h));
        int n = c4 * 4 + e;
        BsHi[n * LDK + r] = h;
        BsLo[n * LDK + r] = l;
      }
    }
  };

  v8f acc[2][2];
#pragma unroll
  for (int mt = 0; mt < 2; ++mt)
#pragma unroll
    for (int nt = 0; nt < 2; ++nt)
#pragma unroll
      for (int e = 0; e < 8; ++e) acc[mt][nt][e] = 0.0f;

  loadA(0); loadB(0);
  for (int kt = 0; kt < kTiles; ++kt) {
    storeA(); storeB();
    __syncthreads();
    if (kt + 1 < kTiles) { loadA(kt + 1); loadB(kt + 1); }

#pragma unroll
    for (int ks = 0; ks < BK; ks += 32) {
      // A fragments (16x32 bf16): lane<16 -> K {ks..ks+7},{ks+16..23}; lane>=16 -> +8
      const int am = waveM * 32 + (lane & 15);
      const int ka = ks + ((lane >> 4) << 3);
      FragBF16 ah[2], al[2];
#pragma unroll
      for (int mt = 0; mt < 2; ++mt) {
        int base = (am + mt * 16) * LDK + ka;
        ah[mt].q[0] = *(const u32x4*)&AsHi[base];
        ah[mt].q[1] = *(const u32x4*)&AsHi[base + 16];
        al[mt].q[0] = *(const u32x4*)&AsLo[base];
        al[mt].q[1] = *(const u32x4*)&AsLo[base + 16];
      }
      // B fragments (32x16 bf16): lane<16 -> K {ks..ks+7},{ks+16..23}; lane>=16 -> +8
      const int bl = waveN * 32 + (lane & 15);
      const int kb = ks + ((lane >> 4) << 3);
#pragma unroll
      for (int nt = 0; nt < 2; ++nt) {
        FragBF16 bh, blo;
        int base = (bl + nt * 16) * LDK + kb;
        bh.q[0]  = *(const u32x4*)&BsHi[base];
        bh.q[1]  = *(const u32x4*)&BsHi[base + 16];
        blo.q[0] = *(const u32x4*)&BsLo[base];
        blo.q[1] = *(const u32x4*)&BsLo[base + 16];
#pragma unroll
        for (int mt = 0; mt < 2; ++mt) {
          // split-bf16: A*B ~= Ahi*Bhi + Ahi*Blo + Alo*Bhi (f32 accumulate)
          acc[mt][nt] = __builtin_amdgcn_wmma_f32_16x16x32_bf16(
              false, al[mt].v, false, bh.v, (short)0, acc[mt][nt], false, false);
          acc[mt][nt] = __builtin_amdgcn_wmma_f32_16x16x32_bf16(
              false, ah[mt].v, false, blo.v, (short)0, acc[mt][nt], false, false);
          acc[mt][nt] = __builtin_amdgcn_wmma_f32_16x16x32_bf16(
              false, ah[mt].v, false, bh.v, (short)0, acc[mt][nt], false, false);
        }
      }
    }
    __syncthreads();
  }

  // Epilogue: C/D layout — lane<16: N=lane, M=v; lane>=16: N=lane-16, M=v+8
  const int col0 = bn + waveN * 32 + (lane & 15);
  const int row0 = bm + waveM * 32 + ((lane >> 4) << 3);
#pragma unroll
  for (int nt = 0; nt < 2; ++nt) {
    int col = col0 + nt * 16;
    float bv = bias[col];
#pragma unroll
    for (int mt = 0; mt < 2; ++mt) {
      int rbase = row0 + mt * 16;
#pragma unroll
      for (int v = 0; v < 8; ++v) {
        float val = acc[mt][nt][v] + bv;
        if (relu) val = fmaxf(val, 0.0f);
        C[(size_t)(rbase + v) * N + col] = val;
        if (hist && val > 0.0f) {
          unsigned b = __float_as_uint(val) >> 20;   // positive floats: monotone key, < 2048
          atomicAdd(&shHist[b], 1u);
        }
      }
    }
  }
  if (hist) {
    __syncthreads();
    for (int i = tid; i < NBINS; i += NTHREADS)
      if (shHist[i]) atomicAdd(&hist[i], shHist[i]);
  }
}

// Single block: suffix-scan histogram from the top, emit bin lower-edge as threshold.
__global__ __launch_bounds__(256)
void topk_threshold_kernel(const unsigned* __restrict__ hist,
                           const int* __restrict__ kPerRow, int rows,
                           float* __restrict__ out)
{
  __shared__ unsigned long long sfx[256];
  const int t = threadIdx.x;
  const unsigned long long ktot =
      (unsigned long long)(*kPerRow) * (unsigned long long)rows;

  unsigned bins[8];
  unsigned long long csum = 0;
#pragma unroll
  for (int i = 0; i < 8; ++i) { bins[i] = hist[t * 8 + i]; csum += bins[i]; }
  sfx[t] = csum;
  __syncthreads();
  for (int off = 1; off < 256; off <<= 1) {
    unsigned long long add = (t + off < 256) ? sfx[t + off] : 0ull;
    __syncthreads();
    sfx[t] += add;
    __syncthreads();
  }
  unsigned long long above = (t + 1 < 256) ? sfx[t + 1] : 0ull;
  if (t == 0 && sfx[0] < ktot) *out = 0.0f;   // fewer positives than k: keep all
  unsigned long long cum = above;
#pragma unroll
  for (int i = 7; i >= 0; --i) {
    cum += bins[i];
    if (cum >= ktot && (cum - bins[i]) < ktot)
      *out = __uint_as_float(((unsigned)(t * 8 + i)) << 20);
  }
}

__global__ void zero_u32_kernel(unsigned* __restrict__ p, int n) {
  int i = blockIdx.x * blockDim.x + threadIdx.x;
  if (i < n) p[i] = 0;
}

extern "C" void kernel_launch(void* const* d_in, const int* in_sizes, int n_in,
                              void* d_out, int out_size, void* d_ws, size_t ws_size,
                              hipStream_t stream) {
  (void)in_sizes; (void)n_in; (void)out_size; (void)ws_size;
  const float* x  = (const float*)d_in[0];
  const float* W1 = (const float*)d_in[1];
  const float* b1 = (const float*)d_in[2];
  const float* W2 = (const float*)d_in[3];
  const float* b2 = (const float*)d_in[4];
  const float* W3 = (const float*)d_in[5];   // W_dec2
  const float* b3 = (const float*)d_in[6];
  const float* W4 = (const float*)d_in[7];   // W_dec1
  const float* b4 = (const float*)d_in[8];
  const int* k_mid  = (const int*)d_in[9];
  const int* k_feat = (const int*)d_in[10];

  const int NR = 4096, DM = 1024, DMID = 4096, DFEAT = 16384;

  char* ws = (char*)d_ws;
  float*    h1   = (float*)ws;                                          // [NR,DMID] (reused for h3)
  float*    h2   = (float*)(ws + (size_t)NR * DMID * sizeof(float));    // [NR,DFEAT]
  unsigned* hist = (unsigned*)(ws + (size_t)NR * DMID * sizeof(float)
                                  + (size_t)NR * DFEAT * sizeof(float)); // 3*2048 u32
  float*    th   = (float*)(hist + 3 * NBINS);                           // 3 floats

  zero_u32_kernel<<<(3 * NBINS + 255) / 256, 256, 0, stream>>>(hist, 3 * NBINS);

  dim3 blk(NTHREADS);
  // 1) h1 = relu(x @ W_enc1 + b1), histogram
  gemm_bias_act_hist_kernel<<<dim3(DMID / BN, NR / BM), blk, 0, stream>>>(
      x, W1, b1, h1, NR, DMID, DM, nullptr, hist + 0 * NBINS, 1);
  topk_threshold_kernel<<<1, 256, 0, stream>>>(hist + 0 * NBINS, k_mid, NR, th + 0);

  // 2) h2 = relu(mask(h1) @ W_enc2 + b2), histogram
  gemm_bias_act_hist_kernel<<<dim3(DFEAT / BN, NR / BM), blk, 0, stream>>>(
      h1, W2, b2, h2, NR, DFEAT, DMID, th + 0, hist + 1 * NBINS, 1);
  topk_threshold_kernel<<<1, 256, 0, stream>>>(hist + 1 * NBINS, k_feat, NR, th + 1);

  // 3) h3 = relu(mask(h2) @ W_dec2 + b3), histogram   (h3 overwrites h1 buffer)
  gemm_bias_act_hist_kernel<<<dim3(DMID / BN, NR / BM), blk, 0, stream>>>(
      h2, W3, b3, h1, NR, DMID, DFEAT, th + 1, hist + 2 * NBINS, 1);
  topk_threshold_kernel<<<1, 256, 0, stream>>>(hist + 2 * NBINS, k_mid, NR, th + 2);

  // 4) recon = mask(h3) @ W_dec1 + b4  (no relu, no histogram)
  gemm_bias_act_hist_kernel<<<dim3(DM / BN, NR / BM), blk, 0, stream>>>(
      h1, W4, b4, (float*)d_out, NR, DM, DMID, th + 2, nullptr, 0);
}
